// TransformerBlock_40312563040384
// MI455X (gfx1250) — compile-verified, matrix operand order checked by reading, not run on hardware
//
#include <hip/hip_runtime.h>
#include <hip/hip_bf16.h>

typedef __bf16 v16bf __attribute__((ext_vector_type(16)));
typedef float  v8f   __attribute__((ext_vector_type(8)));
typedef float  v4f   __attribute__((ext_vector_type(4)));

#define HS 68  // LDS row stride (floats): 272B rows -> 16B-aligned chunks, conflict-free

// ---------------- wave-local LDS sync (CDNA5 split counters) ----------------
__device__ __forceinline__ void wave_lds_sync() {
  asm volatile("" ::: "memory");
#if __has_builtin(__builtin_amdgcn_s_wait_dscnt)
  __builtin_amdgcn_s_wait_dscnt(0);
#else
  asm volatile("s_wait_dscnt 0x0" ::: "memory");
#endif
  asm volatile("" ::: "memory");
  __builtin_amdgcn_wave_barrier();
}

// ---------------- fragment builders ----------------
// B fragment from pre-swizzled LDS: one contiguous 32B read per lane.
__device__ __forceinline__ v16bf load_b_swz(const __bf16* Wlds, int t, int s, int lane) {
  return *(const v16bf*)(Wlds + ((((s << 2) + t) << 5) + lane) * 16);
}

__device__ __forceinline__ v16bf cvt16(v4f a, v4f b, v4f c, v4f d) {
  v16bf r;
#pragma unroll
  for (int i = 0; i < 4; ++i) {
    r[i]      = (__bf16)a[i];
    r[4 + i]  = (__bf16)b[i];
    r[8 + i]  = (__bf16)c[i];
    r[12 + i] = (__bf16)d[i];
  }
  return r;
}

// A fragment from one f32 row (16B-aligned, stride-64 global row).
__device__ __forceinline__ v16bf load_a_row(const float* rowp, int s, int lane) {
  const float* p = rowp + (s << 5) + ((lane >> 4) << 3);
  return cvt16(*(const v4f*)(p), *(const v4f*)(p + 4),
               *(const v4f*)(p + 16), *(const v4f*)(p + 20));
}

// A fragment from the wave's LDS staging buffer (16 x HS f32), plain convert.
__device__ __forceinline__ v16bf load_a_plain(const float* hb, int s, int lane) {
  const float* p = hb + (lane & 15) * HS + (s << 5) + ((lane >> 4) << 3);
  return cvt16(*(const v4f*)(p), *(const v4f*)(p + 4),
               *(const v4f*)(p + 16), *(const v4f*)(p + 20));
}

// A fragment with LayerNorm affine + ReLU applied on the fly (vectorized).
__device__ __forceinline__ v16bf load_a_ln(const float* hb, int s, int lane,
                                           float mean, float rstd,
                                           const float* g, const float* bt) {
  int k0 = (s << 5) + ((lane >> 4) << 3);
  const float* p = hb + (lane & 15) * HS + k0;
  v16bf r;
#pragma unroll
  for (int c = 0; c < 4; ++c) {
    const int off = (c < 2) ? (c << 2) : (8 + (c << 2));   // 0,4,16,20
    v4f x  = *(const v4f*)(p + off);
    v4f gv = *(const v4f*)(g + k0 + off);
    v4f bv = *(const v4f*)(bt + k0 + off);
#pragma unroll
    for (int i = 0; i < 4; ++i) {
      float y = (x[i] - mean) * rstd * gv[i] + bv[i];
      r[(c << 2) + i] = (__bf16)fmaxf(y, 0.0f);
    }
  }
  return r;
}

// Per-row LayerNorm stats over 64 cols: lanes L and L+16 split the row.
__device__ __forceinline__ void ln_stats(const float* hb, int lane, float& mean, float& rstd) {
  const v4f* p = (const v4f*)(hb + (lane & 15) * HS + ((lane >> 4) << 5));
  v4f s4 = {0, 0, 0, 0}, q4 = {0, 0, 0, 0};
#pragma unroll
  for (int i = 0; i < 8; ++i) { v4f v = p[i]; s4 += v; q4 += v * v; }
  float s  = s4[0] + s4[1] + s4[2] + s4[3];
  float s2 = q4[0] + q4[1] + q4[2] + q4[3];
  s  += __shfl_xor(s,  16, 32);
  s2 += __shfl_xor(s2, 16, 32);
  mean = s * 0.015625f;
  float var = s2 * 0.015625f - mean * mean;
  rstd = rsqrtf(fmaxf(var, 0.f) + 1e-5f);
}

#define WMMA_BF16(acc, a, b) \
  (acc) = __builtin_amdgcn_wmma_f32_16x16x32_bf16(false, (a), false, (b), (short)0, (acc), false, false)

// Float atomic max via monotonic integer mapping (works with init = -3e38).
__device__ __forceinline__ void atomicMaxFloat(float* addr, float val) {
  if (val >= 0.0f) atomicMax((int*)addr, __float_as_int(val));
  else             atomicMin((unsigned int*)addr, (unsigned int)__float_as_int(val));
}

// ---------------- weight conversion: f32 [K=64][N=64] -> bf16 B-fragment order
// layout: frag(s,t) [32 lanes][16 elems] contiguous; elem e of lane maps to
// k = 32s + 8*(lane>>4) + (e<8 ? e : e+8), n = 16t + (lane&15)
__global__ void k_swz(const float* __restrict__ W, unsigned short* __restrict__ dst) {
  int i = blockIdx.x * 256 + threadIdx.x;          // [0, 4096)
  int e    = i & 15;
  int lane = (i >> 4) & 31;
  int t    = (i >> 9) & 3;
  int s    = (i >> 11) & 1;
  int k = (s << 5) + (((lane >> 4) & 1) << 3) + ((e < 8) ? e : e + 8);
  int n = (t << 4) + (lane & 15);
  ((__bf16*)dst)[i] = (__bf16)W[k * 64 + n];
}

__global__ void k_init(float* amax, float* den, float* num, long n) {
  long i = (long)blockIdx.x * 256 + threadIdx.x;
  if (i < n) { amax[i] = -3.0e38f; den[i] = 0.0f; num[i] = 0.0f; }
}

// ---------------- xr = relu(x @ W_in + b_in) ----------------
__global__ void __launch_bounds__(256) k_lin_in(const float* __restrict__ x,
                                                const unsigned short* __restrict__ Wg,
                                                const float* __restrict__ bg,
                                                float* __restrict__ out, int Nn) {
  __shared__ __align__(32) __bf16 W[4096];
  __shared__ __align__(16) float  bs[64];
  for (int i = threadIdx.x; i < 4096; i += 256) W[i] = ((const __bf16*)Wg)[i];
  if (threadIdx.x < 64) bs[threadIdx.x] = bg[threadIdx.x];
  __syncthreads();

  int lane = threadIdx.x & 31, wave = threadIdx.x >> 5;
  int lane15 = lane & 15, h2 = lane >> 4;
  int ntile = (Nn + 15) >> 4;
  int nwaves = gridDim.x * 8;

  for (int tile = blockIdx.x * 8 + wave; tile < ntile; tile += nwaves) {
    int row0 = tile << 4;
    int rowA = row0 + lane15; if (rowA >= Nn) rowA = Nn - 1;
    const float* rowp = x + (size_t)rowA * 64;

    v8f z = {0,0,0,0,0,0,0,0};
    v8f acc[4] = {z,z,z,z};
#pragma unroll
    for (int s = 0; s < 2; ++s) {
      v16bf a = load_a_row(rowp, s, lane);
#pragma unroll
      for (int t = 0; t < 4; ++t) WMMA_BF16(acc[t], a, load_b_swz(W, t, s, lane));
    }
    if (row0 + 16 <= Nn) {              // full tile: straight-line stores
#pragma unroll
      for (int t = 0; t < 4; ++t) {
        int col = (t << 4) + lane15; float b = bs[col];
#pragma unroll
        for (int r = 0; r < 8; ++r) {
          int row = row0 + r + (h2 << 3);
          out[(size_t)row * 64 + col] = fmaxf(acc[t][r] + b, 0.0f);
        }
      }
    } else {                            // tail tile: guarded
#pragma unroll
      for (int t = 0; t < 4; ++t) {
        int col = (t << 4) + lane15; float b = bs[col];
#pragma unroll
        for (int r = 0; r < 8; ++r) {
          int row = row0 + r + (h2 << 3);
          if (row < Nn) out[(size_t)row * 64 + col] = fmaxf(acc[t][r] + b, 0.0f);
        }
      }
    }
  }
}

// ---------------- a_src / a_dst / v : three GEMMs sharing the A fragment ----
__global__ void __launch_bounds__(256) k_node_proj(const float* __restrict__ xr,
                                                   const unsigned short* __restrict__ WsG,
                                                   const unsigned short* __restrict__ WdG,
                                                   const unsigned short* __restrict__ WvG,
                                                   float* __restrict__ aS, float* __restrict__ aD,
                                                   float* __restrict__ vV, int Nn) {
  __shared__ __align__(32) __bf16 Ws[4096], Wd[4096], Wv[4096];
  for (int i = threadIdx.x; i < 4096; i += 256) {
    Ws[i] = ((const __bf16*)WsG)[i];
    Wd[i] = ((const __bf16*)WdG)[i];
    Wv[i] = ((const __bf16*)WvG)[i];
  }
  __syncthreads();

  int lane = threadIdx.x & 31, wave = threadIdx.x >> 5;
  int lane15 = lane & 15, h2 = lane >> 4;
  int ntile = (Nn + 15) >> 4;
  int nwaves = gridDim.x * 8;

  for (int tile = blockIdx.x * 8 + wave; tile < ntile; tile += nwaves) {
    int row0 = tile << 4;
    int rowA = row0 + lane15; if (rowA >= Nn) rowA = Nn - 1;
    const float* rowp = xr + (size_t)rowA * 64;

    v8f z = {0,0,0,0,0,0,0,0};
    v8f accS[4] = {z,z,z,z}, accD[4] = {z,z,z,z}, accV[4] = {z,z,z,z};
#pragma unroll
    for (int s = 0; s < 2; ++s) {
      v16bf a = load_a_row(rowp, s, lane);
#pragma unroll
      for (int t = 0; t < 4; ++t) {
        WMMA_BF16(accS[t], a, load_b_swz(Ws, t, s, lane));
        WMMA_BF16(accD[t], a, load_b_swz(Wd, t, s, lane));
        WMMA_BF16(accV[t], a, load_b_swz(Wv, t, s, lane));
      }
    }
    if (row0 + 16 <= Nn) {
#pragma unroll
      for (int t = 0; t < 4; ++t) {
        int col = (t << 4) + lane15;
#pragma unroll
        for (int r = 0; r < 8; ++r) {
          size_t o = (size_t)(row0 + r + (h2 << 3)) * 64 + col;
          aS[o] = accS[t][r]; aD[o] = accD[t][r]; vV[o] = accV[t][r];
        }
      }
    } else {
#pragma unroll
      for (int t = 0; t < 4; ++t) {
        int col = (t << 4) + lane15;
#pragma unroll
        for (int r = 0; r < 8; ++r) {
          int row = row0 + r + (h2 << 3);
          if (row < Nn) {
            size_t o = (size_t)row * 64 + col;
            aS[o] = accS[t][r]; aD[o] = accD[t][r]; vV[o] = accV[t][r];
          }
        }
      }
    }
  }
}

// ---------------- edge kernel: pos-MLP + att-MLP, softmax passes ------------
template <int PHASE>
__global__ void __launch_bounds__(256) k_edge(
    const int* __restrict__ eidx, const float* __restrict__ pos,
    const float* __restrict__ aS, const float* __restrict__ aD, const float* __restrict__ vV,
    const float* __restrict__ posW1, const float* __restrict__ posB1,
    const float* __restrict__ posG,  const float* __restrict__ posBt, const float* __restrict__ posB2,
    const float* __restrict__ attB1, const float* __restrict__ attG,
    const float* __restrict__ attBt, const float* __restrict__ attB2,
    const unsigned short* __restrict__ Wp2G, const unsigned short* __restrict__ Wa1G,
    const unsigned short* __restrict__ Wa2G,
    float* __restrict__ amax, float* __restrict__ den, float* __restrict__ num,
    int E, int Nn) {
  __shared__ __align__(32) __bf16 Wp2[4096], Wa1[4096], Wa2[4096];
  __shared__ __align__(16) float hbuf[8][16 * HS];
  __shared__ __align__(16) float pw1[192];
  __shared__ __align__(16) float pb1[64], pg[64], pbt[64], pb2[64];
  __shared__ __align__(16) float ab1[64], ag[64], abt[64], ab2[64];
  __shared__ int   sidx[8][16], didx[8][16];
  __shared__ float pdl[8][48];

  int tid = threadIdx.x;
  for (int i = tid; i < 4096; i += 256) {
    Wp2[i] = ((const __bf16*)Wp2G)[i];
    Wa1[i] = ((const __bf16*)Wa1G)[i];
    Wa2[i] = ((const __bf16*)Wa2G)[i];
  }
  if (tid < 192) pw1[tid] = posW1[tid];
  if (tid < 64) {
    pb1[tid] = posB1[tid]; pg[tid]  = posG[tid];  pbt[tid] = posBt[tid]; pb2[tid] = posB2[tid];
    ab1[tid] = attB1[tid]; ag[tid]  = attG[tid];  abt[tid] = attBt[tid]; ab2[tid] = attB2[tid];
  }
  __syncthreads();

  int lane = tid & 31, wave = tid >> 5;
  long ET = (long)E + Nn;
  long ntile = (ET + 15) >> 4;
  long nwaves = (long)gridDim.x * 8;

  float* hb  = hbuf[wave];
  int*   si  = sidx[wave];
  int*   di  = didx[wave];
  float* pdw = pdl[wave];
  int lane15 = lane & 15, h2 = lane >> 4;
  int col0   = lane15 << 2;   // 4-wide column chunk owned by this lane

  for (long tile = (long)blockIdx.x * 8 + wave; tile < ntile; tile += nwaves) {
    long e0 = tile << 4;

    // indices + positional diffs for 16 edges (self-loops appended after E)
    if (lane < 16) {
      long ei = e0 + lane;
      int s, d;
      if (ei < (long)E) { s = eidx[ei]; d = eidx[(long)E + ei]; }
      else { long n = ei - E; if (n >= Nn) n = 0; s = (int)n; d = (int)n; }
      si[lane] = s; di[lane] = d;
      pdw[lane * 3 + 0] = pos[(size_t)d * 3 + 0] - pos[(size_t)s * 3 + 0];
      pdw[lane * 3 + 1] = pos[(size_t)d * 3 + 1] - pos[(size_t)s * 3 + 1];
      pdw[lane * 3 + 2] = pos[(size_t)d * 3 + 2] - pos[(size_t)s * 3 + 2];
    }
    wave_lds_sync();

    // pos layer-1 (K=3) in VALU, 4-wide chunks per lane
#pragma unroll
    for (int i = 0; i < 8; ++i) {
      int row = (i << 1) + h2;
      const float* p = &pdw[row * 3];
      float p0 = p[0], p1 = p[1], p2 = p[2];
      v4f b  = *(const v4f*)(pb1 + col0);
      v4f w0 = *(const v4f*)(pw1 + col0);
      v4f w1 = *(const v4f*)(pw1 + 64 + col0);
      v4f w2 = *(const v4f*)(pw1 + 128 + col0);
      *(v4f*)(hb + row * HS + col0) = b + p0 * w0 + p1 * w1 + p2 * w2;
    }
    wave_lds_sync();

    float m1, r1; ln_stats(hb, lane, m1, r1);

    v8f z = {0,0,0,0,0,0,0,0};
    v8f dacc[4] = {z,z,z,z};
#pragma unroll
    for (int s = 0; s < 2; ++s) {
      v16bf a = load_a_ln(hb, s, lane, m1, r1, pg, pbt);
#pragma unroll
      for (int t = 0; t < 4; ++t) WMMA_BF16(dacc[t], a, load_b_swz(Wp2, t, s, lane));
    }
#pragma unroll
    for (int t = 0; t < 4; ++t) {
      float b2 = pb2[(t << 4) + lane15];
#pragma unroll
      for (int r = 0; r < 8; ++r) dacc[t][r] += b2;   // dacc = delta (kept for pass 2)
    }

    // u = delta + (a_dst[d] - a_src[s])  -> staged in LDS
    wave_lds_sync();
#pragma unroll
    for (int t = 0; t < 4; ++t) {
      int col = (t << 4) + lane15;
#pragma unroll
      for (int r = 0; r < 8; ++r) hb[(r + (h2 << 3)) * HS + col] = dacc[t][r];
    }
    wave_lds_sync();
#pragma unroll
    for (int i = 0; i < 8; ++i) {
      int row = (i << 1) + h2;
      int s = si[row], d = di[row];
      v4f qd = *(const v4f*)(aD + (size_t)d * 64 + col0);
      v4f qs = *(const v4f*)(aS + (size_t)s * 64 + col0);
      v4f* hp = (v4f*)(hb + row * HS + col0);
      *hp = *hp + (qd - qs);
    }
    wave_lds_sync();

    // att layer-1
    v8f tacc[4] = {z,z,z,z};
#pragma unroll
    for (int s = 0; s < 2; ++s) {
      v16bf a = load_a_plain(hb, s, lane);
#pragma unroll
      for (int t = 0; t < 4; ++t) WMMA_BF16(tacc[t], a, load_b_swz(Wa1, t, s, lane));
    }
    wave_lds_sync();
#pragma unroll
    for (int t = 0; t < 4; ++t) {
      int col = (t << 4) + lane15; float b1v = ab1[col];
#pragma unroll
      for (int r = 0; r < 8; ++r) hb[(r + (h2 << 3)) * HS + col] = tacc[t][r] + b1v;
    }
    wave_lds_sync();

    // att LN+ReLU then layer-2 -> alpha
    float m2, r2; ln_stats(hb, lane, m2, r2);
    v8f aacc[4] = {z,z,z,z};
#pragma unroll
    for (int s = 0; s < 2; ++s) {
      v16bf a = load_a_ln(hb, s, lane, m2, r2, ag, abt);
#pragma unroll
      for (int t = 0; t < 4; ++t) WMMA_BF16(aacc[t], a, load_b_swz(Wa2, t, s, lane));
    }
#pragma unroll
    for (int t = 0; t < 4; ++t) {
      float b2 = ab2[(t << 4) + lane15];
#pragma unroll
      for (int r = 0; r < 8; ++r) aacc[t][r] += b2;
    }

    int dr[8], srr[8];
#pragma unroll
    for (int r = 0; r < 8; ++r) {
      int row = r + (h2 << 3);
      dr[r] = di[row]; srr[r] = si[row];
    }
    bool full = (e0 + 16 <= ET);        // wave-uniform tail test

    if (PHASE == 1) {
      if (full) {
#pragma unroll
        for (int t = 0; t < 4; ++t) {
          int col = (t << 4) + lane15;
#pragma unroll
          for (int r = 0; r < 8; ++r)
            atomicMaxFloat(&amax[(size_t)dr[r] * 64 + col], aacc[t][r]);
        }
      } else {
#pragma unroll
        for (int t = 0; t < 4; ++t) {
          int col = (t << 4) + lane15;
#pragma unroll
          for (int r = 0; r < 8; ++r)
            if (e0 + r + (h2 << 3) < ET)
              atomicMaxFloat(&amax[(size_t)dr[r] * 64 + col], aacc[t][r]);
        }
      }
    } else {
      if (full) {
#pragma unroll
        for (int t = 0; t < 4; ++t) {
          int col = (t << 4) + lane15;
#pragma unroll
          for (int r = 0; r < 8; ++r) {
            size_t o = (size_t)dr[r] * 64 + col;
            float ee = __expf(aacc[t][r] - amax[o]);
            atomicAdd(&den[o], ee);
            float val = vV[(size_t)srr[r] * 64 + col] + dacc[t][r];
            atomicAdd(&num[o], ee * val);
          }
        }
      } else {
#pragma unroll
        for (int t = 0; t < 4; ++t) {
          int col = (t << 4) + lane15;
#pragma unroll
          for (int r = 0; r < 8; ++r)
            if (e0 + r + (h2 << 3) < ET) {
              size_t o = (size_t)dr[r] * 64 + col;
              float ee = __expf(aacc[t][r] - amax[o]);
              atomicAdd(&den[o], ee);
              float val = vV[(size_t)srr[r] * 64 + col] + dacc[t][r];
              atomicAdd(&num[o], ee * val);
            }
        }
      }
    }
    wave_lds_sync();   // hbuf/index buffers reused next iteration
  }
}

// ---------------- out = relu((num/denom) @ W_out + b_out) -------------------
__global__ void __launch_bounds__(256) k_out(const float* __restrict__ num,
                                             const float* __restrict__ den,
                                             const unsigned short* __restrict__ Wg,
                                             const float* __restrict__ bg,
                                             float* __restrict__ out, int Nn) {
  __shared__ __align__(32) __bf16 W[4096];
  __shared__ __align__(16) float  bs[64];
  for (int i = threadIdx.x; i < 4096; i += 256) W[i] = ((const __bf16*)Wg)[i];
  if (threadIdx.x < 64) bs[threadIdx.x] = bg[threadIdx.x];
  __syncthreads();

  int lane = threadIdx.x & 31, wave = threadIdx.x >> 5;
  int lane15 = lane & 15, h2 = lane >> 4;
  int ntile = (Nn + 15) >> 4;
  int nwaves = gridDim.x * 8;

  for (int tile = blockIdx.x * 8 + wave; tile < ntile; tile += nwaves) {
    int row0 = tile << 4;
    int rowA = row0 + lane15; if (rowA >= Nn) rowA = Nn - 1;
    const float* nrow = num + (size_t)rowA * 64;
    const float* drow = den + (size_t)rowA * 64;

    v8f z = {0,0,0,0,0,0,0,0};
    v8f acc[4] = {z,z,z,z};
#pragma unroll
    for (int s = 0; s < 2; ++s) {
      int k0 = (s << 5) + ((lane >> 4) << 3);
      v16bf a;
#pragma unroll
      for (int c = 0; c < 4; ++c) {
        const int off = (c < 2) ? (c << 2) : (8 + (c << 2));  // 0,4,16,20
        v4f nv = *(const v4f*)(nrow + k0 + off);
        v4f dv = *(const v4f*)(drow + k0 + off);
#pragma unroll
        for (int i = 0; i < 4; ++i)
          a[(c << 2) + i] = (__bf16)(nv[i] / fmaxf(dv[i], 1e-16f));
      }
#pragma unroll
      for (int t = 0; t < 4; ++t) WMMA_BF16(acc[t], a, load_b_swz(W, t, s, lane));
    }
    if (row0 + 16 <= Nn) {
#pragma unroll
      for (int t = 0; t < 4; ++t) {
        int col = (t << 4) + lane15; float b = bs[col];
#pragma unroll
        for (int r = 0; r < 8; ++r) {
          int row = row0 + r + (h2 << 3);
          out[(size_t)row * 64 + col] = fmaxf(acc[t][r] + b, 0.0f);
        }
      }
    } else {
#pragma unroll
      for (int t = 0; t < 4; ++t) {
        int col = (t << 4) + lane15; float b = bs[col];
#pragma unroll
        for (int r = 0; r < 8; ++r) {
          int row = row0 + r + (h2 << 3);
          if (row < Nn) out[(size_t)row * 64 + col] = fmaxf(acc[t][r] + b, 0.0f);
        }
      }
    }
  }
}

// ---------------- host side ----------------
static inline int imin(int a, int b) { return a < b ? a : b; }

extern "C" void kernel_launch(void* const* d_in, const int* in_sizes, int n_in,
                              void* d_out, int out_size, void* d_ws, size_t ws_size,
                              hipStream_t stream) {
  const float* x      = (const float*)d_in[0];
  const float* pos    = (const float*)d_in[1];
  const int*   eidx   = (const int*)d_in[2];
  const float* W_in   = (const float*)d_in[3];
  const float* b_in   = (const float*)d_in[4];
  const float* W_lin  = (const float*)d_in[5];
  const float* W_src  = (const float*)d_in[6];
  const float* W_dst  = (const float*)d_in[7];
  const float* pos_W1 = (const float*)d_in[8];
  const float* pos_b1 = (const float*)d_in[9];
  const float* pos_g  = (const float*)d_in[10];
  const float* pos_bt = (const float*)d_in[11];
  const float* pos_W2 = (const float*)d_in[12];
  const float* pos_b2 = (const float*)d_in[13];
  const float* att_W1 = (const float*)d_in[14];
  const float* att_b1 = (const float*)d_in[15];
  const float* att_g  = (const float*)d_in[16];
  const float* att_bt = (const float*)d_in[17];
  const float* att_W2 = (const float*)d_in[18];
  const float* att_b2 = (const float*)d_in[19];
  const float* W_out  = (const float*)d_in[20];
  const float* b_out  = (const float*)d_in[21];

  int Nn = in_sizes[0] / 64;
  int E  = in_sizes[2] / 2;
  size_t NC = (size_t)Nn * 64;

  float* ws   = (float*)d_ws;
  float* xr   = ws;
  float* aS   = xr + NC;
  float* aD   = aS + NC;
  float* vV   = aD + NC;
  float* amax = vV + NC;
  float* den  = amax + NC;
  float* num  = den + NC;
  unsigned short* wb = (unsigned short*)(num + NC);
  unsigned short* bW_in  = wb;
  unsigned short* bW_src = wb + 1 * 4096;
  unsigned short* bW_dst = wb + 2 * 4096;
  unsigned short* bW_lin = wb + 3 * 4096;
  unsigned short* bW_p2  = wb + 4 * 4096;
  unsigned short* bW_a1  = wb + 5 * 4096;
  unsigned short* bW_a2  = wb + 6 * 4096;
  unsigned short* bW_out = wb + 7 * 4096;

  k_swz<<<16, 256, 0, stream>>>(W_in,   bW_in);
  k_swz<<<16, 256, 0, stream>>>(W_src,  bW_src);
  k_swz<<<16, 256, 0, stream>>>(W_dst,  bW_dst);
  k_swz<<<16, 256, 0, stream>>>(W_lin,  bW_lin);
  k_swz<<<16, 256, 0, stream>>>(pos_W2, bW_p2);
  k_swz<<<16, 256, 0, stream>>>(att_W1, bW_a1);
  k_swz<<<16, 256, 0, stream>>>(att_W2, bW_a2);
  k_swz<<<16, 256, 0, stream>>>(W_out,  bW_out);

  int initBlk = (int)((NC + 255) / 256);
  k_init<<<initBlk, 256, 0, stream>>>(amax, den, num, (long)NC);

  int ntile = (Nn + 15) / 16;
  int nblk  = imin((ntile + 7) / 8, 512);     // persistent waves, >= several tiles each
  k_lin_in<<<nblk, 256, 0, stream>>>(x, bW_in, b_in, xr, Nn);
  k_node_proj<<<nblk, 256, 0, stream>>>(xr, bW_src, bW_dst, bW_lin, aS, aD, vV, Nn);

  long ET = (long)E + Nn;
  long etile = (ET + 15) / 16;
  int  eblk  = imin((int)((etile + 7) / 8), 2048);  // amortize 24KB weight staging
  k_edge<1><<<eblk, 256, 0, stream>>>(eidx, pos, aS, aD, vV,
      pos_W1, pos_b1, pos_g, pos_bt, pos_b2,
      att_b1, att_g, att_bt, att_b2,
      bW_p2, bW_a1, bW_a2, amax, den, num, E, Nn);
  k_edge<2><<<eblk, 256, 0, stream>>>(eidx, pos, aS, aD, vV,
      pos_W1, pos_b1, pos_g, pos_bt, pos_b2,
      att_b1, att_g, att_bt, att_b2,
      bW_p2, bW_a1, bW_a2, amax, den, num, E, Nn);

  k_out<<<nblk, 256, 0, stream>>>(num, den, bW_out, b_out, (float*)d_out, Nn);
}
